// DeltaNet_31877247271533
// MI455X (gfx1250) — compile-verified, matrix operand order checked by reading, not run on hardware
//
#include <hip/hip_runtime.h>
#include <hip/hip_bf16.h>
#include <cstdint>
#include <cstddef>

typedef __attribute__((ext_vector_type(16))) __bf16 v16bf;
typedef __attribute__((ext_vector_type(4)))  __bf16 bf16x4;
typedef __attribute__((ext_vector_type(2)))  __bf16 bf16x2;
typedef __attribute__((ext_vector_type(8)))  float  v8f;

#define WMMA_BF16(a, b, c) \
  __builtin_amdgcn_wmma_f32_16x16x32_bf16(false, (a), false, (b), (short)0, (c), false, false)

// ---------------------------------------------------------------------------
// WMMA fragment helpers (wave32), layouts per CDNA5 ISA §7.12.2.
// ---------------------------------------------------------------------------
__device__ __forceinline__ int kpat(int j, int hf) {
  return (j < 4) ? (hf * 8 + j * 2) : (16 + hf * 8 + (j - 4) * 2);
}

// fragment from bf16 LDS: lane walks FIRST dim, K pairs contiguous in SECOND.
__device__ __forceinline__ v16bf fragA_lds(const __bf16* src, int ld, int m0, int k0) {
  int lane = (int)(threadIdx.x & 31);
  int m = m0 + (lane & 15);
  int hf = lane >> 4;
  v16bf a;
#pragma unroll
  for (int j = 0; j < 8; ++j) {
    int kk = k0 + kpat(j, hf);
    bf16x2 p = *(const bf16x2*)(src + m * ld + kk);
    a[2 * j] = p[0];
    a[2 * j + 1] = p[1];
  }
  return a;
}

// bf16 fragment from [k][n] storage (pairs strided by ld) — cold paths only.
__device__ __forceinline__ v16bf fragB_lds(const __bf16* src, int ld, int k0, int n0) {
  int lane = (int)(threadIdx.x & 31);
  int n = n0 + (lane & 15);
  int hf = lane >> 4;
  v16bf b;
#pragma unroll
  for (int j = 0; j < 8; ++j) {
    int kk = k0 + kpat(j, hf);
    b[2 * j] = src[kk * ld + n];
    b[2 * j + 1] = src[(kk + 1) * ld + n];
  }
  return b;
}

// fp32 source variant (pairs contiguous), converts on load (T, u^T, S^T).
__device__ __forceinline__ v16bf fragA_rm(const float* src, int ld, int m0, int k0) {
  int lane = (int)(threadIdx.x & 31);
  int m = m0 + (lane & 15);
  int hf = lane >> 4;
  v16bf a;
#pragma unroll
  for (int j = 0; j < 8; ++j) {
    int kk = k0 + kpat(j, hf);
    a[2 * j] = (__bf16)src[m * ld + kk];
    a[2 * j + 1] = (__bf16)src[m * ld + kk + 1];
  }
  return a;
}

__device__ __forceinline__ void storeC_bf16(__bf16* dst, int ld, int m0, int n0, v8f c) {
  int lane = (int)(threadIdx.x & 31);
  int n = n0 + (lane & 15);
  int mb = m0 + ((lane >> 4) ? 8 : 0);
#pragma unroll
  for (int r = 0; r < 8; ++r) dst[(mb + r) * ld + n] = (__bf16)c[r];
}

// C tile stored transposed: dst[n*ld + m]; per-lane writes contiguous in m.
__device__ __forceinline__ void storeC_T(float* dst, int ld, int m0, int n0, v8f c) {
  int lane = (int)(threadIdx.x & 31);
  int n = n0 + (lane & 15);
  int mb = m0 + ((lane >> 4) ? 8 : 0);
#pragma unroll
  for (int r = 0; r < 8; ++r) dst[n * ld + mb + r] = c[r];
}

__device__ __forceinline__ v8f loadC_T(const float* src, int ld, int m0, int n0) {
  int lane = (int)(threadIdx.x & 31);
  int n = n0 + (lane & 15);
  int mb = m0 + ((lane >> 4) ? 8 : 0);
  v8f c;
#pragma unroll
  for (int r = 0; r < 8; ++r) c[r] = src[n * ld + mb + r];
  return c;
}

__device__ __forceinline__ float silu_f(float x) { return x / (1.f + __expf(-x)); }

// CDNA5 async global->LDS copy: 16B per lane, tracked by ASYNCcnt.
__device__ __forceinline__ void async_load16(unsigned ldsByte, const float* gptr) {
  asm volatile("global_load_async_to_lds_b128 %0, %1, off"
               :: "v"(ldsByte), "v"(gptr) : "memory");
}
__device__ __forceinline__ void wait_async0() {
  asm volatile("s_wait_asynccnt 0x0" ::: "memory");
}

// ---------------------------------------------------------------------------
// Generic GEMM: C[M,N] = act(A[M,K] @ B[K,N] + bias), bf16 WMMA, f32 accum.
// 256 threads (8 waves), block tile 64x128, K-step 32. Each wave owns a 32x32
// output (4 accumulators, 4 WMMAs per K-step). bf16 staged in LDS with B
// transposed; full-tile fast path uses b128 global loads + b64 LDS stores.
// ---------------------------------------------------------------------------
__global__ void __launch_bounds__(256)
gemm_bias_act(const float* __restrict__ A, const float* __restrict__ Bm,
              float* __restrict__ C, int M, int N, int K,
              const float* __restrict__ bias, int act) {
  __shared__ __attribute__((aligned(16))) __bf16 As[64 * 36];
  __shared__ __attribute__((aligned(16))) __bf16 BsT[128 * 36];  // [n][k]
  int tid = (int)threadIdx.x;
  int lane = tid & 31, wid = tid >> 5;
  int m0 = (int)blockIdx.y * 64, n0 = (int)blockIdx.x * 128;
  int wy = wid >> 2, wx = wid & 3;  // wave tile: rows wy*32..+31, cols wx*32..+31
  v8f acc00 = {0.f, 0.f, 0.f, 0.f, 0.f, 0.f, 0.f, 0.f};
  v8f acc01 = acc00, acc10 = acc00, acc11 = acc00;
  int KT = (K + 31) >> 5;
  for (int kt = 0; kt < KT; ++kt) {
    int k0 = kt << 5;
    bool full = (m0 + 64 <= M) && (n0 + 128 <= N) && (k0 + 32 <= K);
    if (full) {
      // A tile 64x32: float4 loads, bf16x4 stores (512 units, 2 per thread)
#pragma unroll
      for (int i = tid; i < 512; i += 256) {
        int r = i >> 3, c4 = (i & 7) * 4;
        const float4 x = *(const float4*)(A + (size_t)(m0 + r) * K + k0 + c4);
        bf16x4 t;
        t[0] = (__bf16)x.x; t[1] = (__bf16)x.y;
        t[2] = (__bf16)x.z; t[3] = (__bf16)x.w;
        *(bf16x4*)(As + r * 36 + c4) = t;
      }
      // B tile 32x128 -> BsT[n][k]: unit = k-pair x 4n (512 units)
#pragma unroll
      for (int i = tid; i < 512; i += 256) {
        int kp = i >> 5, ng = i & 31;
        int kk = k0 + 2 * kp, nn = n0 + 4 * ng;
        const float4 r0 = *(const float4*)(Bm + (size_t)kk * N + nn);
        const float4 r1 = *(const float4*)(Bm + (size_t)(kk + 1) * N + nn);
        bf16x2 t;
        t[0] = (__bf16)r0.x; t[1] = (__bf16)r1.x;
        *(bf16x2*)(BsT + (4 * ng + 0) * 36 + 2 * kp) = t;
        t[0] = (__bf16)r0.y; t[1] = (__bf16)r1.y;
        *(bf16x2*)(BsT + (4 * ng + 1) * 36 + 2 * kp) = t;
        t[0] = (__bf16)r0.z; t[1] = (__bf16)r1.z;
        *(bf16x2*)(BsT + (4 * ng + 2) * 36 + 2 * kp) = t;
        t[0] = (__bf16)r0.w; t[1] = (__bf16)r1.w;
        *(bf16x2*)(BsT + (4 * ng + 3) * 36 + 2 * kp) = t;
      }
    } else {
      for (int i = tid; i < 64 * 16; i += 256) {  // A pairs (guarded)
        int r = i >> 4, cp = (i & 15) * 2;
        int gm = m0 + r, gk = k0 + cp;
        float x0 = (gm < M && gk < K) ? A[(size_t)gm * K + gk] : 0.f;
        float x1 = (gm < M && gk + 1 < K) ? A[(size_t)gm * K + gk + 1] : 0.f;
        bf16x2 t; t[0] = (__bf16)x0; t[1] = (__bf16)x1;
        *(bf16x2*)(As + r * 36 + cp) = t;
      }
      for (int i = tid; i < 128 * 16; i += 256) {  // B^T pairs (guarded)
        int n = i >> 4, cp = (i & 15) * 2;
        int gn = n0 + n, gk = k0 + cp;
        float x0 = (gk < K && gn < N) ? Bm[(size_t)gk * N + gn] : 0.f;
        float x1 = (gk + 1 < K && gn < N) ? Bm[(size_t)(gk + 1) * N + gn] : 0.f;
        bf16x2 t; t[0] = (__bf16)x0; t[1] = (__bf16)x1;
        *(bf16x2*)(BsT + n * 36 + cp) = t;
      }
    }
    __syncthreads();
    v16bf a0 = fragA_lds(As, 36, wy * 32, 0);
    v16bf a1 = fragA_lds(As, 36, wy * 32 + 16, 0);
    v16bf b0 = fragA_lds(BsT, 36, wx * 32, 0);
    v16bf b1 = fragA_lds(BsT, 36, wx * 32 + 16, 0);
    acc00 = WMMA_BF16(a0, b0, acc00);
    acc01 = WMMA_BF16(a0, b1, acc01);
    acc10 = WMMA_BF16(a1, b0, acc10);
    acc11 = WMMA_BF16(a1, b1, acc11);
    __syncthreads();
  }
  int nloc = lane & 15;
  int mhalf = (lane >> 4) ? 8 : 0;
#pragma unroll
  for (int tm = 0; tm < 2; ++tm) {
#pragma unroll
    for (int tn = 0; tn < 2; ++tn) {
      v8f acc = tm == 0 ? (tn == 0 ? acc00 : acc01) : (tn == 0 ? acc10 : acc11);
      int mb = m0 + wy * 32 + tm * 16 + mhalf;
      int gn = n0 + wx * 32 + tn * 16 + nloc;
      if (gn >= N) continue;
      float bv = bias ? bias[gn] : 0.f;
#pragma unroll
      for (int r = 0; r < 8; ++r) {
        int gm = mb + r;
        if (gm >= M) continue;
        float v = acc[r] + bv;
        if (act) v = silu_f(v);
        C[(size_t)gm * N + gn] = v;
      }
    }
  }
}

// ---------------------------------------------------------------------------
__global__ void conv4_silu(const float* __restrict__ x, const float* __restrict__ w,
                           float* __restrict__ y, int BL, int L, int D) {
  int t = (int)(blockIdx.x * blockDim.x + threadIdx.x);
  if (t >= BL * D) return;
  int row = t / D, c = t - row * D;
  int l = row & (L - 1);
  float acc = 0.f;
#pragma unroll
  for (int tt = 0; tt < 4; ++tt) {
    int lt = l - 3 + tt;
    if (lt >= 0) acc += x[(size_t)(row - 3 + tt) * D + c] * w[c * 4 + tt];
  }
  y[t] = silu_f(acc);
}

__global__ void beta_sigmoid(const float* __restrict__ X, const float* __restrict__ Wb,
                             float* __restrict__ beta, int BL, int D, int H) {
  int t = (int)(blockIdx.x * blockDim.x + threadIdx.x);
  if (t >= BL * H) return;
  int row = t / H, h = t - row * H;
  float acc = 0.f;
  for (int d = 0; d < D; ++d) acc += X[(size_t)row * D + d] * Wb[d * H + h];
  beta[t] = 1.f / (1.f + __expf(-acc));
}

// ---------------------------------------------------------------------------
// Chunkwise delta rule. One block per (b, h, dv-slice of 64). S^T (64x256 fp32)
// resident in LDS; q/k chunk tiles fetched with async global->LDS; all matmuls
// bf16 WMMA with fp32 accumulation.
// ---------------------------------------------------------------------------
__global__ void __launch_bounds__(256)
delta_chunk(const float* __restrict__ q, const float* __restrict__ k,
            const float* __restrict__ v, const float* __restrict__ beta,
            float* __restrict__ out, int B, int H, int L) {
  extern __shared__ __attribute__((aligned(16))) char ldsraw[];
  float*  ST  = (float*)ldsraw;        // [64][256]  S^T (dv-major)
  float*  stQ = ST + 64 * 256;         // [32][256]  raw q staging
  float*  stK = stQ + 32 * 256;        // [32][256]  raw k staging
  float*  ubT = stK + 32 * 256;        // [64][32]   u^T -> u_i^T (fp32)
  float*  Tm  = ubT + 64 * 32;         // [32][32]   fp32
  float*  red = Tm + 32 * 32;          // 512
  float*  rnq = red + 512;             // 32
  float*  rnk = rnq + 32;              // 32
  float*  btS = rnk + 32;              // 32
  __bf16* qcb = (__bf16*)(btS + 32);   // [32][256]
  __bf16* kcb = qcb + 32 * 256;        // [32][256]
  __bf16* kcT = kcb + 32 * 256;        // [256][32]  k^T
  __bf16* kbb = kcT + 256 * 32;        // [32][256]  k*beta
  __bf16* wb  = kbb + 32 * 256;        // [32][256]
  __bf16* vcT = wb + 32 * 256;         // [64][32]   (v*beta)^T
  __bf16* Ab  = vcT + 64 * 32;         // [32][32]   attn

  int tid = (int)threadIdx.x;
  int lane = tid & 31, wid = tid >> 5;
  int bid = (int)blockIdx.x;
  int bi = bid >> 4;
  int h  = (bid >> 2) & 3;
  int sl = bid & 3;
  int vof = sl * 64;
  const size_t RS = 1024;  // H * dh
  const v8f vz = {0.f, 0.f, 0.f, 0.f, 0.f, 0.f, 0.f, 0.f};

  for (int i = tid; i < 64 * 256; i += 256) ST[i] = 0.f;
  __syncthreads();

  unsigned bq = (unsigned)(uintptr_t)(void*)stQ;
  unsigned bk = (unsigned)(uintptr_t)(void*)stK;

  int nchunks = L / 32;
  for (int nc = 0; nc < nchunks; ++nc) {
    int l0 = nc * 32;
    const float* qbase = q + (size_t)(bi * L + l0) * RS + h * 256;
    const float* kbase = k + (size_t)(bi * L + l0) * RS + h * 256;
    // async copy 32x256 fp32 q and k tiles into LDS (16B units per lane)
    for (int u = tid; u < 2048; u += 256) {
      int r = u >> 6, cg = (u & 63) * 4;
      async_load16(bq + (unsigned)u * 16, qbase + (size_t)r * RS + cg);
      async_load16(bk + (unsigned)u * 16, kbase + (size_t)r * RS + cg);
    }
    if (tid < 32) btS[tid] = beta[(size_t)(bi * L + l0 + tid) * H + h];
    wait_async0();
    __syncthreads();

    {  // l2norm row sums (8 partials per row)
      int r = tid >> 3, seg = tid & 7;
      float sq = 0.f, sk = 0.f;
      for (int j = 0; j < 32; ++j) {
        float aq = stQ[r * 256 + seg * 32 + j]; sq += aq * aq;
        float ak = stK[r * 256 + seg * 32 + j]; sk += ak * ak;
      }
      red[tid] = sq; red[256 + tid] = sk;
    }
    __syncthreads();
    if (tid < 32) {
      float s1 = 0.f, s2 = 0.f;
      for (int s = 0; s < 8; ++s) { s1 += red[tid * 8 + s]; s2 += red[256 + tid * 8 + s]; }
      rnq[tid] = 1.f / sqrtf(s1 + 1e-12f);
      rnk[tid] = 1.f / sqrtf(s2 + 1e-12f);
    }
    __syncthreads();
    // normalize + bf16 convert once; also build k^T copy
    for (int i = tid; i < 32 * 128; i += 256) {
      int r = i >> 7, cp = (i & 127) * 2;
      float sq = rnq[r], sk = rnk[r], bb = btS[r];
      float q0 = stQ[r * 256 + cp] * sq, q1 = stQ[r * 256 + cp + 1] * sq;
      float k0 = stK[r * 256 + cp] * sk, k1 = stK[r * 256 + cp + 1] * sk;
      bf16x2 t;
      t[0] = (__bf16)q0; t[1] = (__bf16)q1; *(bf16x2*)(qcb + r * 256 + cp) = t;
      t[0] = (__bf16)k0; t[1] = (__bf16)k1; *(bf16x2*)(kcb + r * 256 + cp) = t;
      t[0] = (__bf16)(k0 * bb); t[1] = (__bf16)(k1 * bb);
      *(bf16x2*)(kbb + r * 256 + cp) = t;
      kcT[cp * 32 + r] = (__bf16)k0;
      kcT[(cp + 1) * 32 + r] = (__bf16)k1;
    }
    // v slice: scale by beta, store transposed bf16
    for (int i = tid; i < 32 * 64; i += 256) {
      int r = i >> 6, c = i & 63;
      float vv = v[(size_t)(bi * L + l0 + r) * RS + h * 256 + vof + c] * btS[r];
      vcT[c * 32 + r] = (__bf16)vv;
    }
    __syncthreads();

    // T = -(kb @ kc^T), strict lower
    if (wid < 4) {
      int ti = wid >> 1, tj = wid & 1;
      v8f acc = vz;
      for (int kt = 0; kt < 8; ++kt) {
        v16bf a  = fragA_lds(kbb, 256, ti * 16, kt * 32);
        v16bf bt = fragA_lds(kcb, 256, tj * 16, kt * 32);  // kc^T as B
        acc = WMMA_BF16(a, bt, acc);
      }
      int nn = tj * 16 + (lane & 15);
      int mb = ti * 16 + ((lane >> 4) ? 8 : 0);
#pragma unroll
      for (int r = 0; r < 8; ++r) {
        int mm = mb + r;
        Tm[mm * 32 + nn] = (mm > nn) ? -acc[r] : 0.f;
      }
    }
    __syncthreads();
    // forward substitution + identity (wave 0, LDS ops in-order per wave)
    if (wid == 0) {
      for (int i = 1; i < 32; ++i) {
        float s = 0.f;
        for (int l2 = 0; l2 < 32; ++l2) s += Tm[i * 32 + l2] * Tm[l2 * 32 + lane];
        if (lane < i) Tm[i * 32 + lane] += s;
      }
      Tm[lane * 32 + lane] += 1.f;
    }
    __syncthreads();

    // u = T @ vc (store u^T fp32); w = T @ kb (store bf16)
    {
      int tm = wid >> 2, tn = wid & 3;
      v16bf a = fragA_rm(Tm, 32, tm * 16, 0);
      v16bf bv = fragA_lds(vcT, 32, tn * 16, 0);
      v8f acc = vz;
      acc = WMMA_BF16(a, bv, acc);
      storeC_T(ubT, 32, tm * 16, tn * 16, acc);
#pragma unroll
      for (int t4 = 0; t4 < 4; ++t4) {
        int tnw = (wid & 3) + 4 * t4;
        v16bf bkf = fragB_lds(kbb, 256, 0, tnw * 16);
        v8f accw = vz;
        accw = WMMA_BF16(a, bkf, accw);
        storeC_bf16(wb, 256, tm * 16, tnw * 16, accw);
      }
    }
    __syncthreads();

    // attn = tril_incl(qc @ kc^T), bf16
    if (wid < 4) {
      int ti = wid >> 1, tj = wid & 1;
      v8f acc = vz;
      for (int kt = 0; kt < 8; ++kt) {
        v16bf a  = fragA_lds(qcb, 256, ti * 16, kt * 32);
        v16bf bt = fragA_lds(kcb, 256, tj * 16, kt * 32);
        acc = WMMA_BF16(a, bt, acc);
      }
      int nn = tj * 16 + (lane & 15);
      int mb = ti * 16 + ((lane >> 4) ? 8 : 0);
#pragma unroll
      for (int r = 0; r < 8; ++r) {
        int mm = mb + r;
        Ab[mm * 32 + nn] = (mm >= nn) ? (__bf16)acc[r] : (__bf16)0.f;
      }
    }
    __syncthreads();

    // u_i = u - w @ S  (update u^T in place)
    {
      int tm = wid >> 2, tn = wid & 3;
      v8f acc = vz;
      for (int kt = 0; kt < 8; ++kt) {
        v16bf a  = fragA_lds(wb, 256, tm * 16, kt * 32);
        v16bf bs = fragA_rm(ST, 256, tn * 16, kt * 32);  // S^T: lane->dv
        acc = WMMA_BF16(a, bs, acc);
      }
      int nn = tn * 16 + (lane & 15);
      int mb = tm * 16 + ((lane >> 4) ? 8 : 0);
#pragma unroll
      for (int r = 0; r < 8; ++r) {
        int mm = mb + r;
        ubT[nn * 32 + mm] -= acc[r];
      }
    }
    __syncthreads();

    // o = qc @ S + attn @ u_i -> global
    {
      int tm = wid >> 2, tn = wid & 3;
      v8f acc = vz;
      for (int kt = 0; kt < 8; ++kt) {
        v16bf a  = fragA_lds(qcb, 256, tm * 16, kt * 32);
        v16bf bs = fragA_rm(ST, 256, tn * 16, kt * 32);
        acc = WMMA_BF16(a, bs, acc);
      }
      v16bf a2 = fragA_lds(Ab, 32, tm * 16, 0);
      v16bf b2 = fragA_rm(ubT, 32, tn * 16, 0);  // u_i^T: lane->dv
      acc = WMMA_BF16(a2, b2, acc);
      int nn = tn * 16 + (lane & 15);
      int mb = tm * 16 + ((lane >> 4) ? 8 : 0);
#pragma unroll
      for (int r = 0; r < 8; ++r) {
        int mm = mb + r;
        out[(size_t)(bi * L + l0 + mm) * RS + h * 256 + vof + nn] = acc[r];
      }
    }
    __syncthreads();  // all S reads done before update

    // S += kc^T @ u_i (64 tiles, 8 per wave), accumulate fp32 in S^T
    for (int rep = 0; rep < 8; ++rep) {
      int tile = wid * 8 + rep;
      int tm = tile >> 2, tn = tile & 3;
      v8f acc = loadC_T(ST, 256, tm * 16, tn * 16);
      v16bf a  = fragA_lds(kcT, 32, tm * 16, 0);
      v16bf bu = fragA_rm(ubT, 32, tn * 16, 0);
      acc = WMMA_BF16(a, bu, acc);
      storeC_T(ST, 256, tm * 16, tn * 16, acc);
    }
    __syncthreads();
  }
}

// ---------------------------------------------------------------------------
__global__ void __launch_bounds__(256)
paths_feats(const float* __restrict__ X, const float* __restrict__ v,
            const float* __restrict__ dlt, const float* __restrict__ lw,
            const float* __restrict__ mw, float* __restrict__ localo,
            float* __restrict__ mido, float* __restrict__ rf, int L) {
  __shared__ float sred[14 * 256];
  int bid = (int)blockIdx.x;
  int h = bid & 3;
  int row = bid >> 2;
  int l = row & (L - 1);
  int c = (int)threadIdx.x;
  int ch = h * 256 + c;
  float o0 = 0.f;
#pragma unroll
  for (int t = 0; t < 7; ++t) {
    int lt = l - 6 + t;
    if (lt >= 0) o0 += v[(size_t)(row - 6 + t) * 1024 + ch] * lw[ch * 7 + t];
  }
  float o1 = 0.f;
  for (int t = 0; t < 25; ++t) {
    int lt = l - 24 + t;
    if (lt >= 0) o1 += v[(size_t)(row - 24 + t) * 1024 + ch] * mw[ch * 25 + t];
  }
  float o2 = dlt[(size_t)row * 1024 + ch];
  float o3 = v[(size_t)row * 1024 + ch];
  localo[(size_t)row * 1024 + ch] = o0;
  mido[(size_t)row * 1024 + ch]   = o1;
  rf[(size_t)row * 1080 + ch] = X[(size_t)row * 1024 + ch];
  float vals[14] = {o0, o0 * o0, o1, o1 * o1, o2, o2 * o2, o3, o3 * o3,
                    o0 * o1, o0 * o2, o0 * o3, o1 * o2, o1 * o3, o2 * o3};
#pragma unroll
  for (int f = 0; f < 14; ++f) sred[f * 256 + c] = vals[f];
  __syncthreads();
  for (int off = 128; off > 0; off >>= 1) {
    if (c < off)
      for (int f = 0; f < 14; ++f) sred[f * 256 + c] += sred[f * 256 + c + off];
    __syncthreads();
  }
  if (c == 0) {
    float feats[14];
#pragma unroll
    for (int pth = 0; pth < 4; ++pth) {
      float m = sred[(2 * pth) * 256] / 256.f;
      float s = sred[(2 * pth + 1) * 256];
      feats[2 * pth] = m;
      feats[2 * pth + 1] = (s - 256.f * m * m) / 255.f;
    }
    for (int d = 0; d < 6; ++d) feats[8 + d] = sred[(8 + d) * 256] / 256.f;
    for (int f = 0; f < 14; ++f)
      rf[(size_t)row * 1080 + 1024 + f * 4 + h] = feats[f];
  }
}

__global__ void router_logits(const float* __restrict__ h1, const float* __restrict__ W2,
                              const float* __restrict__ b2, float* __restrict__ p,
                              int BL, int Kd) {
  int t = (int)(blockIdx.x * blockDim.x + threadIdx.x);
  if (t >= BL * 4) return;
  int row = t >> 2, h = t & 3;
  float lg0 = b2[h * 4 + 0], lg1 = b2[h * 4 + 1], lg2 = b2[h * 4 + 2], lg3 = b2[h * 4 + 3];
  for (int j = 0; j < Kd; ++j) {
    float hv = h1[(size_t)row * Kd + j];
    const float* wr = W2 + (size_t)j * 16 + h * 4;
    lg0 += hv * wr[0]; lg1 += hv * wr[1]; lg2 += hv * wr[2]; lg3 += hv * wr[3];
  }
  float mx = fmaxf(fmaxf(lg0, lg1), fmaxf(lg2, lg3));
  float e0 = __expf(lg0 - mx), e1 = __expf(lg1 - mx), e2 = __expf(lg2 - mx), e3 = __expf(lg3 - mx);
  float sc = (1.f - 0.04f) / (e0 + e1 + e2 + e3);
  float* pp = p + (size_t)t * 4;
  pp[0] = e0 * sc + 0.01f; pp[1] = e1 * sc + 0.01f;
  pp[2] = e2 * sc + 0.01f; pp[3] = e3 * sc + 0.01f;
}

__global__ void __launch_bounds__(256)
mix_norm(const float* __restrict__ p, const float* __restrict__ localo,
         const float* __restrict__ mido, const float* __restrict__ dlt,
         const float* __restrict__ v, const float* __restrict__ mixw,
         const float* __restrict__ onw, float* __restrict__ outn) {
  __shared__ float sred[256];
  __shared__ float sval;
  int bid = (int)blockIdx.x;
  int h = bid & 3;
  int row = bid >> 2;
  int c = (int)threadIdx.x;
  int ch = h * 256 + c;
  size_t g = (size_t)row * 1024 + ch;
  const float* pp = p + (size_t)row * 16 + h * 4;
  float mixed = pp[0] * localo[g] + pp[1] * mido[g] + pp[2] * dlt[g] + pp[3] * v[g];
  sred[c] = mixed * mixed;
  __syncthreads();
  for (int off = 128; off > 0; off >>= 1) {
    if (c < off) sred[c] += sred[c + off];
    __syncthreads();
  }
  if (c == 0) sval = sqrtf(sred[0] / 256.f + 1e-5f);
  __syncthreads();
  float m2 = mixed / sval * mixw[ch];
  __syncthreads();
  sred[c] = m2 * m2;
  __syncthreads();
  for (int off = 128; off > 0; off >>= 1) {
    if (c < off) sred[c] += sred[c + off];
    __syncthreads();
  }
  if (c == 0) sval = sqrtf(sred[0] / 256.f + 1e-5f);
  __syncthreads();
  outn[g] = m2 / sval * onw[c];
}

// ---------------------------------------------------------------------------
extern "C" void kernel_launch(void* const* d_in, const int* in_sizes, int n_in,
                              void* d_out, int out_size, void* d_ws, size_t ws_size,
                              hipStream_t stream) {
  (void)in_sizes; (void)n_in; (void)out_size; (void)ws_size;
  const int B = 4, L = 4096, D = 1024, H = 4;
  const int BL = B * L;
  const int RFD = D + 14 * H;      // 1080
  const int HD1 = 2 * RFD;         // 2160

  const float* X    = (const float*)d_in[0];
  const float* Wq   = (const float*)d_in[1];
  const float* Wk   = (const float*)d_in[2];
  const float* Wv   = (const float*)d_in[3];
  const float* Wb   = (const float*)d_in[4];
  const float* cqw  = (const float*)d_in[5];
  const float* ckw  = (const float*)d_in[6];
  const float* cvw  = (const float*)d_in[7];
  const float* lw   = (const float*)d_in[8];
  const float* mw   = (const float*)d_in[9];
  const float* rw1  = (const float*)d_in[10];
  const float* rb1  = (const float*)d_in[11];
  const float* rw2  = (const float*)d_in[12];
  const float* rb2  = (const float*)d_in[13];
  const float* mixw = (const float*)d_in[14];
  const float* onw  = (const float*)d_in[15];
  const float* Wo   = (const float*)d_in[16];
  float* OUT = (float*)d_out;

  char* ws = (char*)d_ws;
  size_t off = 0;
  auto alloc = [&](size_t bytes) -> float* {
    float* ptr = (float*)(ws + off);
    off += (bytes + 255) & ~(size_t)255;
    return ptr;
  };
  float* ws0  = alloc((size_t)BL * D * 4);       // lin -> delta_out
  float* wsQ  = alloc((size_t)BL * D * 4);       // q -> local_out
  float* wsK  = alloc((size_t)BL * D * 4);       // k -> mid_out
  float* wsV  = alloc((size_t)BL * D * 4);       // v
  float* wsRF = alloc((size_t)BL * RFD * 4);     // router_feat -> o_n
  float* wsH1 = alloc((size_t)BL * HD1 * 4);     // h1
  float* wsB  = alloc((size_t)BL * H * 4);       // beta
  float* wsP  = alloc((size_t)BL * H * 4 * 4);   // probs

  dim3 blk(256);
  dim3 gemmGridD((D + 127) / 128, (BL + 63) / 64);

  gemm_bias_act<<<gemmGridD, blk, 0, stream>>>(X, Wq, ws0, BL, D, D, nullptr, 0);
  conv4_silu<<<(BL * D + 255) / 256, blk, 0, stream>>>(ws0, cqw, wsQ, BL, L, D);
  gemm_bias_act<<<gemmGridD, blk, 0, stream>>>(X, Wk, ws0, BL, D, D, nullptr, 0);
  conv4_silu<<<(BL * D + 255) / 256, blk, 0, stream>>>(ws0, ckw, wsK, BL, L, D);
  gemm_bias_act<<<gemmGridD, blk, 0, stream>>>(X, Wv, ws0, BL, D, D, nullptr, 0);
  conv4_silu<<<(BL * D + 255) / 256, blk, 0, stream>>>(ws0, cvw, wsV, BL, L, D);

  beta_sigmoid<<<(BL * H + 255) / 256, blk, 0, stream>>>(X, Wb, wsB, BL, D, H);

  // delta rule: fp32 region 36448 floats + bf16 region 44032 elems
  size_t deltaLds = (size_t)(64*256 + 32*256*2 + 64*32 + 32*32 + 512 + 96) * 4 +
                    (size_t)(32*256*5 + 64*32 + 32*32) * 2;
  delta_chunk<<<dim3(B * H * 4), blk, deltaLds, stream>>>(wsQ, wsK, wsV, wsB, ws0, B, H, L);

  paths_feats<<<dim3(BL * H), blk, 0, stream>>>(X, wsV, ws0, lw, mw, wsQ, wsK, wsRF, L);

  dim3 gemmGridR((HD1 + 127) / 128, (BL + 63) / 64);
  gemm_bias_act<<<gemmGridR, blk, 0, stream>>>(wsRF, rw1, wsH1, BL, HD1, RFD, rb1, 1);

  router_logits<<<(BL * H + 255) / 256, blk, 0, stream>>>(wsH1, rw2, rb2, wsP, BL, HD1);

  mix_norm<<<dim3(BL * H), blk, 0, stream>>>(wsP, wsQ, wsK, ws0, wsV, mixw, onw, wsRF);

  gemm_bias_act<<<gemmGridD, blk, 0, stream>>>(wsRF, Wo, OUT, BL, D, D, nullptr, 0);
}
